// DeformConvNet_24352464568557
// MI455X (gfx1250) — matrix operand order repaired and checked by reading, not compiled, not run on hardware
//
#include <hip/hip_runtime.h>
#include <hip/hip_bf16.h>

typedef __attribute__((ext_vector_type(16))) _Float16 v16h;
typedef __attribute__((ext_vector_type(8)))  float    v8f;

// ---------------------------------------------------------------------------
// Weight pre-pack: OIHW f32 -> f16 A-fragment layout for V_WMMA_F32_16X16X32_F16
// flat index = (((ktile*9 + t)*NCH + q)*32 + lane)*16 + e
// lane: M = lane&15, half = lane>>4 ; e: v=e>>1, pos=e&1
// K-in-chunk kk = half*16 + e ; c = q*32 + kk
// ---------------------------------------------------------------------------
__global__ __launch_bounds__(256) void prepack_kernel(const float* __restrict__ w,
                                                      _Float16* __restrict__ out,
                                                      int Kout, int Cin) {
    int idx = blockIdx.x * 256 + threadIdx.x;
    int total = Kout * Cin * 9;
    if (idx >= total) return;
    int e    = idx & 15;
    int lane = (idx >> 4) & 31;
    int rest = idx >> 9;                 // (ktile*9 + t)*NCH + q
    int NCH  = Cin >> 5;
    int q    = rest % NCH;
    int tt   = rest / NCH;
    int t    = tt % 9;
    int ktile = tt / 9;
    int half = lane >> 4;
    int m    = lane & 15;
    int v    = e >> 1, pos = e & 1;
    int kk   = half * 16 + e;
    int c    = q * 32 + kk;
    int k    = ktile * 16 + m;
    out[idx] = (_Float16)w[(size_t)(k * Cin + c) * 9 + t];
}

// ---------------------------------------------------------------------------
// conv11: 1->32 channels, 3x3 pad1, bias + relu, B=32, H=W=128 (tiny FLOPs)
// ---------------------------------------------------------------------------
__global__ __launch_bounds__(256) void conv11_kernel(const float* __restrict__ x,
                                                     const float* __restrict__ w,
                                                     const float* __restrict__ bias,
                                                     float* __restrict__ out) {
    size_t idx = (size_t)blockIdx.x * 256 + threadIdx.x; // 32*32*128*128
    int xo = idx & 127;
    int yo = (idx >> 7) & 127;
    int k  = (idx >> 14) & 31;
    int bb = idx >> 19;
    const float* xp = x + (size_t)bb * 16384;
    float s = bias[k];
#pragma unroll
    for (int r = 0; r < 3; ++r)
#pragma unroll
        for (int c = 0; c < 3; ++c) {
            int yy = yo + r - 1, xx = xo + c - 1;
            if (yy >= 0 && yy < 128 && xx >= 0 && xx < 128)
                s += w[k * 9 + r * 3 + c] * xp[yy * 128 + xx];
        }
    out[idx] = fmaxf(s, 0.0f);
}

// ---------------------------------------------------------------------------
// Implicit-GEMM conv via WMMA, halo-padded NHWC f16 input -> no bounds checks.
// Each wave computes a 16(M) x 64(N) tile: 4 accumulators share each A frag.
// Software-pipelined one K-step ahead so VMEM overlaps the matrix pipe.
// Input layout: (B, HIN+2, WIN+2, CIN), border pre-zeroed (= conv zero pad).
// ---------------------------------------------------------------------------
template <int CIN, int HIN, int WIN, int STRIDE, bool BIASRELU>
__global__ __launch_bounds__(128) void gemm_conv_kernel(
        const _Float16* __restrict__ in_nhwc,
        const _Float16* __restrict__ apack,
        const float*    __restrict__ bias,
        float*          __restrict__ out_nchw,  // (B, Kout, HOUT, WOUT)
        int Kout) {
    constexpr int HOUT = HIN / STRIDE;
    constexpr int WOUT = WIN / STRIDE;
    constexpr int WP   = WIN + 2;
    constexpr int HP   = HIN + 2;
    constexpr int NCH  = CIN / 32;
    constexpr int KSTEPS = 9 * NCH;

    const int lane   = threadIdx.x;               // 0..31
    const int half   = lane >> 4;
    const int nl     = lane & 15;
    const int ktile  = blockIdx.y;
    const int ngroup = blockIdx.x * 4 + threadIdx.y;   // group of 64 pixels

    const _Float16* bptr[4];
    int xo4[4], yo4[4], bb4[4];
#pragma unroll
    for (int i = 0; i < 4; ++i) {
        int pix = ngroup * 64 + i * 16 + nl;
        int xo  = pix % WOUT;
        int t1  = pix / WOUT;
        int yo  = t1 % HOUT;
        int bb  = t1 / HOUT;
        xo4[i] = xo; yo4[i] = yo; bb4[i] = bb;
        bptr[i] = in_nhwc +
                  ((size_t)(bb * HP + yo * STRIDE) * WP + xo * STRIDE) * CIN +
                  half * 16;
    }

    v8f acc0 = {}, acc1 = {}, acc2 = {}, acc3 = {};
    const _Float16* ap = apack + (size_t)ktile * 9 * NCH * 512 + lane * 16;

    // prologue: load K-step 0 fragments
    v16h a_c  = *(const v16h*)ap;
    ap += 512;
    v16h b0c = *(const v16h*)(bptr[0]);
    v16h b1c = *(const v16h*)(bptr[1]);
    v16h b2c = *(const v16h*)(bptr[2]);
    v16h b3c = *(const v16h*)(bptr[3]);

#pragma unroll
    for (int step = 0; step < KSTEPS; ++step) {
        v16h a_n, b0n, b1n, b2n, b3n;
        if (step + 1 < KSTEPS) {           // constant-folded after unroll
            const int s2 = step + 1;
            const int t2 = s2 / NCH, q2 = s2 % NCH;
            const int off = ((t2 / 3) * WP + (t2 % 3)) * CIN + q2 * 32;
            a_n = *(const v16h*)ap;
            ap += 512;
            b0n = *(const v16h*)(bptr[0] + off);
            b1n = *(const v16h*)(bptr[1] + off);
            b2n = *(const v16h*)(bptr[2] + off);
            b3n = *(const v16h*)(bptr[3] + off);
        }
        acc0 = __builtin_amdgcn_wmma_f32_16x16x32_f16(false, a_c, false, b0c, (short)0, acc0, false, false);
        acc1 = __builtin_amdgcn_wmma_f32_16x16x32_f16(false, a_c, false, b1c, (short)0, acc1, false, false);
        acc2 = __builtin_amdgcn_wmma_f32_16x16x32_f16(false, a_c, false, b2c, (short)0, acc2, false, false);
        acc3 = __builtin_amdgcn_wmma_f32_16x16x32_f16(false, a_c, false, b3c, (short)0, acc3, false, false);
        if (step + 1 < KSTEPS) {
            a_c = a_n; b0c = b0n; b1c = b1n; b2c = b2n; b3c = b3n;
        }
    }

    const int kbase = ktile * 16 + 8 * half;  // C/D layout: VGPR r -> M = r + 8*half
#pragma unroll
    for (int i = 0; i < 4; ++i) {
        const v8f acc = (i == 0) ? acc0 : (i == 1) ? acc1 : (i == 2) ? acc2 : acc3;
#pragma unroll
        for (int r = 0; r < 8; ++r) {
            int k = kbase + r;
            float v = acc[r];
            if (BIASRELU) { v += bias[k]; v = fmaxf(v, 0.0f); }
            out_nchw[(((size_t)bb4[i] * Kout + k) * HOUT + yo4[i]) * WOUT + xo4[i]] = v;
        }
    }
}

// ---------------------------------------------------------------------------
// Deterministic two-stage per-channel BN statistics (sum, sumsq)
// ---------------------------------------------------------------------------
__global__ __launch_bounds__(256) void reduce_stats_kernel(
        const float* __restrict__ x, float* __restrict__ partial,
        int C, int HW, int B, int S) {
    const int c = blockIdx.x;
    const int nelem = B * HW;
    float s = 0.f, s2 = 0.f;
    for (int j = blockIdx.y * 256 + threadIdx.x; j < nelem; j += S * 256) {
        int b = j / HW, p = j % HW;
        float v = x[((size_t)b * C + c) * HW + p];
        s += v; s2 += v * v;
    }
    __shared__ float sh[512];
    sh[threadIdx.x] = s;
    sh[256 + threadIdx.x] = s2;
    __syncthreads();
    for (int st = 128; st > 0; st >>= 1) {
        if (threadIdx.x < st) {
            sh[threadIdx.x] += sh[threadIdx.x + st];
            sh[256 + threadIdx.x] += sh[256 + threadIdx.x + st];
        }
        __syncthreads();
    }
    if (threadIdx.x == 0) {
        partial[((size_t)c * S + blockIdx.y) * 2 + 0] = sh[0];
        partial[((size_t)c * S + blockIdx.y) * 2 + 1] = sh[256];
    }
}

__global__ __launch_bounds__(256) void finalize_stats_kernel(
        const float* __restrict__ partial, float* __restrict__ stats, int C, int S) {
    int c = blockIdx.x * 256 + threadIdx.x;
    if (c >= C) return;
    float s = 0.f, s2 = 0.f;
    for (int i = 0; i < S; ++i) {
        s  += partial[((size_t)c * S + i) * 2 + 0];
        s2 += partial[((size_t)c * S + i) * 2 + 1];
    }
    stats[c]     = s;
    stats[C + c] = s2;
}

// ---------------------------------------------------------------------------
// BN normalize in place (f32 NCHW) + emit f16 copy into padded NHWC buffer
// ---------------------------------------------------------------------------
__global__ __launch_bounds__(256) void bn_normalize_kernel(
        float* __restrict__ x, const float* __restrict__ stats,
        const float* __restrict__ g, const float* __restrict__ b,
        _Float16* __restrict__ out_nhwc, int C, int H, int W, int B) {
    const int HW = H * W;
    size_t idx = (size_t)blockIdx.x * 256 + threadIdx.x;
    size_t total = (size_t)B * C * HW;
    if (idx >= total) return;
    int p  = idx % HW;
    int c  = (idx / HW) % C;
    int bb = idx / ((size_t)HW * C);
    float cnt  = (float)B * (float)HW;
    float mean = stats[c] / cnt;
    float var  = stats[C + c] / cnt - mean * mean;
    float inv  = rsqrtf(var + 1e-5f);
    float v    = g[c] * (x[idx] - mean) * inv + b[c];
    x[idx] = v;
    int y = p / W, xp = p % W;
    out_nhwc[((size_t)(bb * (H + 2) + y + 1) * (W + 2) + xp + 1) * C + c] = (_Float16)v;
}

// ---------------------------------------------------------------------------
// Deform gather: faithful to reference's contiguous (b*c, h*w, 2) view:
// for pixel j, (dy,dx) = plane[2j], plane[2j+1] where plane spans channels
// 2c and 2c+1 of the NCHW offset tensor (adjacent in memory).
// Writes the bilinear sample as f16 into the padded NHWC buffer.
// ---------------------------------------------------------------------------
__global__ __launch_bounds__(256) void deform_gather_kernel(
        const float* __restrict__ h, const float* __restrict__ off,
        _Float16* __restrict__ d_nhwc, int C, int H, int W, int B) {
    size_t idx = (size_t)blockIdx.x * 256 + threadIdx.x;
    const int HW = H * W;
    size_t total = (size_t)B * C * HW;
    if (idx >= total) return;
    int j  = idx % HW;
    int c  = (idx / HW) % C;
    int bb = idx / ((size_t)HW * C);
    const float* op = off + ((size_t)bb * 2 * C + 2 * c) * HW;
    float dy = op[2 * j];
    float dx = op[2 * j + 1];
    int jy = j / W, jx = j % W;
    float cy = fminf(fmaxf(dy + (float)jy, 0.0f), (float)(H - 1));
    float cx = fminf(fmaxf(dx + (float)jx, 0.0f), (float)(W - 1));
    float y0f = floorf(cy), y1f = ceilf(cy);
    float x0f = floorf(cx), x1f = ceilf(cx);
    float ty = cy - y0f, tx = cx - x0f;
    int y0 = (int)y0f, y1 = (int)y1f, x0 = (int)x0f, x1 = (int)x1f;
    const float* hp = h + ((size_t)bb * C + c) * HW;
    float v_lt = hp[y0 * W + x0];
    float v_rb = hp[y1 * W + x1];
    float v_lb = hp[y0 * W + x1];
    float v_rt = hp[y1 * W + x0];
    float vt = ty * (v_rt - v_lt) + v_lt;
    float vb = ty * (v_rb - v_lb) + v_lb;
    float v  = tx * (vb - vt) + vt;
    d_nhwc[((size_t)(bb * (H + 2) + jy + 1) * (W + 2) + jx + 1) * C + c] = (_Float16)v;
}

// ---------------------------------------------------------------------------
// Head: BN (inlined) + global average pool + FC + softmax. One block per batch.
// ---------------------------------------------------------------------------
__global__ __launch_bounds__(128) void head_kernel(
        const float* __restrict__ raw4, const float* __restrict__ stats,
        const float* __restrict__ g, const float* __restrict__ b,
        const float* __restrict__ fcw, const float* __restrict__ fcb,
        float* __restrict__ out) {
    const int bb = blockIdx.x;
    const int c  = threadIdx.x;          // 128 channels
    const float cnt = 32.0f * 1024.0f;
    float mean = stats[c] / cnt;
    float var  = stats[128 + c] / cnt - mean * mean;
    float inv  = rsqrtf(var + 1e-5f);
    const float* p = raw4 + ((size_t)bb * 128 + c) * 1024;
    float s = 0.f;
    for (int i = 0; i < 1024; ++i) s += p[i];
    float pooled = g[c] * (s * (1.0f / 1024.0f) - mean) * inv + b[c];
    __shared__ float sm[128];
    __shared__ float logits[10];
    sm[c] = pooled;
    __syncthreads();
    if (c < 10) {
        float l = fcb[c];
        for (int k = 0; k < 128; ++k) l += sm[k] * fcw[c * 128 + k];
        logits[c] = l;
    }
    __syncthreads();
    if (c == 0) {
        float mx = logits[0];
        for (int i = 1; i < 10; ++i) mx = fmaxf(mx, logits[i]);
        float e[10]; float se = 0.f;
        for (int i = 0; i < 10; ++i) { e[i] = expf(logits[i] - mx); se += e[i]; }
        for (int i = 0; i < 10; ++i) out[bb * 10 + i] = e[i] / se;
    }
}

// ---------------------------------------------------------------------------
// Orchestration
// ---------------------------------------------------------------------------
extern "C" void kernel_launch(void* const* d_in, const int* in_sizes, int n_in,
                              void* d_out, int out_size, void* d_ws, size_t ws_size,
                              hipStream_t stream) {
    const float* x        = (const float*)d_in[0];
    const float* conv11_w = (const float*)d_in[1];
    const float* conv11_b = (const float*)d_in[2];
    const float* bn11_g   = (const float*)d_in[3];
    const float* bn11_b   = (const float*)d_in[4];
    const float* off12_w  = (const float*)d_in[5];
    const float* conv12_w = (const float*)d_in[6];
    const float* conv12_b = (const float*)d_in[7];
    const float* bn12_g   = (const float*)d_in[8];
    const float* bn12_b   = (const float*)d_in[9];
    const float* off21_w  = (const float*)d_in[10];
    const float* conv21_w = (const float*)d_in[11];
    const float* conv21_b = (const float*)d_in[12];
    const float* bn21_g   = (const float*)d_in[13];
    const float* bn21_b   = (const float*)d_in[14];
    const float* off22_w  = (const float*)d_in[15];
    const float* conv22_w = (const float*)d_in[16];
    const float* conv22_b = (const float*)d_in[17];
    const float* bn22_g   = (const float*)d_in[18];
    const float* bn22_b   = (const float*)d_in[19];
    const float* fc_w     = (const float*)d_in[20];
    const float* fc_b     = (const float*)d_in[21];

    const size_t REG = 134217728ULL; // 128 MiB per ping-pong region
    char* base = (char*)d_ws;
    float*     X    = (float*)(base);                 // f32 NCHW region
    _Float16*  Yh   = (_Float16*)(base + REG);        // padded f16 NHWC region
    float*     Z    = (float*)(base + 2 * REG);       // f32 NCHW region
    _Float16*  pack = (_Float16*)(base + 3 * REG);    // 2 MiB weight packs
    float*     st   = (float*)(base + 3 * REG + (2u << 20));
    float* s1 = st;
    float* s2 = st + 512;
    float* s3 = st + 1024;
    float* s4 = st + 1536;
    float* part = st + 2048; // up to 128*64*2 floats

    _Float16* p_off12  = pack + 0;
    _Float16* p_conv12 = pack + 18432;
    _Float16* p_off21  = pack + 36864;
    _Float16* p_conv21 = pack + 110592;
    _Float16* p_off22  = pack + 184320;
    _Float16* p_conv22 = pack + 479232;

    auto nb = [](long n) { return (unsigned)((n + 255) / 256); };
    const int S = 64;

    // padded f16 buffer sizes (bytes)
    const size_t PAD_L1 = (size_t)32 * 130 * 130 * 32 * 2;   // h1f16 / d1
    const size_t PAD_L2 = (size_t)32 * 66 * 66 * 64 * 2;     // h2f16 / d2
    const size_t PAD_L3 = (size_t)32 * 66 * 66 * 128 * 2;    // h3f16 / d3

    // --- weight pre-pack (all six WMMA convs) ---
    prepack_kernel<<<nb(64L * 32 * 9),   256, 0, stream>>>(off12_w,  p_off12,  64, 32);
    prepack_kernel<<<nb(64L * 32 * 9),   256, 0, stream>>>(conv12_w, p_conv12, 64, 32);
    prepack_kernel<<<nb(128L * 64 * 9),  256, 0, stream>>>(off21_w,  p_off21,  128, 64);
    prepack_kernel<<<nb(128L * 64 * 9),  256, 0, stream>>>(conv21_w, p_conv21, 128, 64);
    prepack_kernel<<<nb(256L * 128 * 9), 256, 0, stream>>>(off22_w,  p_off22,  256, 128);
    prepack_kernel<<<nb(128L * 128 * 9), 256, 0, stream>>>(conv22_w, p_conv22, 128, 128);

    // --- stage 1: conv11 + relu -> X ; BN -> h1 (X in place) + h1f16 (Yh padded) ---
    conv11_kernel<<<nb(32L * 32 * 128 * 128), 256, 0, stream>>>(x, conv11_w, conv11_b, X);
    reduce_stats_kernel<<<dim3(32, S), 256, 0, stream>>>(X, part, 32, 16384, 32, S);
    finalize_stats_kernel<<<1, 256, 0, stream>>>(part, s1, 32, S);
    hipMemsetAsync(Yh, 0, PAD_L1, stream);
    bn_normalize_kernel<<<nb(32L * 32 * 16384), 256, 0, stream>>>(X, s1, bn11_g, bn11_b, Yh, 32, 128, 128, 32);

    // --- stage 2: off12 GEMM -> Z ; gather -> Yh ; conv12 GEMM -> X ; BN ---
    gemm_conv_kernel<32, 128, 128, 1, false><<<dim3(2048, 4), dim3(32, 4), 0, stream>>>(Yh, p_off12, nullptr, Z, 64);
    deform_gather_kernel<<<nb(32L * 32 * 16384), 256, 0, stream>>>(X, Z, Yh, 32, 128, 128, 32);
    gemm_conv_kernel<32, 128, 128, 2, true><<<dim3(512, 4), dim3(32, 4), 0, stream>>>(Yh, p_conv12, conv12_b, X, 64);
    reduce_stats_kernel<<<dim3(64, S), 256, 0, stream>>>(X, part, 64, 4096, 32, S);
    finalize_stats_kernel<<<1, 256, 0, stream>>>(part, s2, 64, S);
    hipMemsetAsync(Yh, 0, PAD_L2, stream);
    bn_normalize_kernel<<<nb(32L * 64 * 4096), 256, 0, stream>>>(X, s2, bn12_g, bn12_b, Yh, 64, 64, 64, 32);

    // --- stage 3: off21 GEMM -> Z ; gather -> Yh ; conv21 GEMM -> Z ; BN ---
    gemm_conv_kernel<64, 64, 64, 1, false><<<dim3(512, 8), dim3(32, 4), 0, stream>>>(Yh, p_off21, nullptr, Z, 128);
    deform_gather_kernel<<<nb(32L * 64 * 4096), 256, 0, stream>>>(X, Z, Yh, 64, 64, 64, 32);
    gemm_conv_kernel<64, 64, 64, 1, true><<<dim3(512, 8), dim3(32, 4), 0, stream>>>(Yh, p_conv21, conv21_b, Z, 128);
    reduce_stats_kernel<<<dim3(128, S), 256, 0, stream>>>(Z, part, 128, 4096, 32, S);
    finalize_stats_kernel<<<1, 256, 0, stream>>>(part, s3, 128, S);
    hipMemsetAsync(Yh, 0, PAD_L3, stream);
    bn_normalize_kernel<<<nb(32L * 128 * 4096), 256, 0, stream>>>(Z, s3, bn21_g, bn21_b, Yh, 128, 64, 64, 32);

    // --- stage 4: off22 GEMM -> X ; gather -> Yh ; conv22 GEMM -> X ; BN+head ---
    gemm_conv_kernel<128, 64, 64, 1, false><<<dim3(512, 16), dim3(32, 4), 0, stream>>>(Yh, p_off22, nullptr, X, 256);
    deform_gather_kernel<<<nb(32L * 128 * 4096), 256, 0, stream>>>(Z, X, Yh, 128, 64, 64, 32);
    gemm_conv_kernel<128, 64, 64, 2, true><<<dim3(128, 8), dim3(32, 4), 0, stream>>>(Yh, p_conv22, conv22_b, X, 128);
    reduce_stats_kernel<<<dim3(128, S), 256, 0, stream>>>(X, part, 128, 1024, 32, S);
    finalize_stats_kernel<<<1, 256, 0, stream>>>(part, s4, 128, S);
    head_kernel<<<32, 128, 0, stream>>>(X, s4, bn22_g, bn22_b, fc_w, fc_b, (float*)d_out);
}
